// WaveletConv_9706626090086
// MI455X (gfx1250) — matrix operand order repaired and checked by reading, not compiled, not run on hardware
//
#include <hip/hip_runtime.h>

typedef __attribute__((ext_vector_type(16))) _Float16 v16h;
typedef __attribute__((ext_vector_type(8)))  float    v8f;
typedef unsigned int u32x4 __attribute__((ext_vector_type(4)));
typedef int          i32x4 __attribute__((ext_vector_type(4)));
typedef int          i32x8 __attribute__((ext_vector_type(8)));

#define L_KERNEL   251
#define K_PAD      256      // K padded for 8 chunks of 32
#define W_STRIDE   264      // LDS row stride (halves): 528B != 0 mod 256B -> no bank conflicts
#define N_OUT      64
#define BATCH      32
#define N_SAMPLES  32768
#define PADL       125
#define TILE_T     256

// ---------------------------------------------------------------------------
// Kernel 1: build the scaled wavelet bank (64 x 256 f16, zero padded k>=251).
// Mirrors _scale_wavelets: nearest interpolate, center, divide by sqrt(a).
// ---------------------------------------------------------------------------
__global__ void build_bank(const float* __restrict__ a,
                           const float* __restrict__ mother,
                           _Float16* __restrict__ wbank) {
    const int c = blockIdx.x;        // channel 0..63
    const int k = threadIdx.x;       // tap 0..255
    const float A_MIN = fmaxf(5.0f / 500.0f, 11.0f / 251.0f);
    const float a_c   = fminf(fmaxf(a[c], A_MIN), 1.0f);
    const int l_out   = (int)floorf((float)L_KERNEL * a_c);
    const int offset  = (L_KERNEL - l_out) / 2;

    float val = 0.0f;
    if (k < L_KERNEL) {
        const int p = k - offset;
        if (p >= 0 && p < l_out) {
            int src = (p * L_KERNEL) / l_out;
            src = min(max(src, 0), L_KERNEL - 1);
            val = mother[src] / sqrtf(a_c);
        }
    }
    wbank[c * K_PAD + k] = (_Float16)val;
}

// ---------------------------------------------------------------------------
// Kernel 2: implicit-GEMM wavelet convolution via v_wmma_f32_16x16x32_f16.
//   D[m, n] = sum_k W[m, k] * xs[n + k],  xs[j] = x[tbase - 125 + j]
// Block: 256 threads (8 waves). Wave w: M-tile = w%4, N-tiles = (w/4)*8 .. +7.
// Bank staged to LDS by the Tensor Data Mover; TDM pad feature inserts the
// 4-dword row gap (256 -> 264 halves) that avoids LDS bank conflicts.
// ---------------------------------------------------------------------------
__global__ __launch_bounds__(256) void wavelet_conv_wmma(
        const float* __restrict__ x,
        const _Float16* __restrict__ wbank,
        float* __restrict__ out) {
    __shared__ _Float16 lds_w[N_OUT * W_STRIDE];          // 64 x 264 halves
    __shared__ _Float16 xs[TILE_T + K_PAD + 8];           // 520 halves

    const int tid   = threadIdx.x;
    const int b     = blockIdx.y;
    const int tbase = blockIdx.x * TILE_T;

    // --- TDM: DMA the 32KB f16 bank into LDS (wave 0 issues, all waves sync).
    // Descriptor: 1-D tile of 8192 dwords; pad_enable with pad_interval=6
    // (128 dwords) and pad_amount=3 (4 dwords) -> 264-half LDS row stride.
    if (tid < 32) {
        const unsigned lds_off = (unsigned)(size_t)(void*)lds_w;   // LDS byte addr (low 32 bits)
        const unsigned long long ga = (unsigned long long)(size_t)wbank;
        const u32x4 g0 = {
            1u,                                            // count=1 (valid user D#)
            lds_off,                                       // lds_addr
            (unsigned)(ga & 0xFFFFFFFFu),                  // global_addr[31:0]
            (unsigned)((ga >> 32) & 0x01FFFFFFu) | 0x80000000u  // addr[56:32] | type=2
        };
        const i32x8 g1 = {
            (2 << 16) | (1 << 20) | (6 << 22) | (3 << 25), // data_size=4B, pad_en, intv=128dw, amt=4dw
            (int)((8192u & 0xFFFFu) << 16),                // tensor_dim0 lo16 -> bits[63:48]
            (int)((8192u >> 16) | (1u << 16)),             // dim0 hi16, tensor_dim1=1
            (int)(8192u << 16),                            // tile_dim0 = 8192 dwords
            1,                                             // tile_dim1 = 1
            8192,                                          // tensor_dim0_stride lo32
            0, 0
        };
        const i32x4 gz4 = { 0, 0, 0, 0 };
        const i32x8 gz8 = { 0, 0, 0, 0, 0, 0, 0, 0 };
        __builtin_amdgcn_tensor_load_to_lds(g0, g1, gz4, gz4, gz8, 0);
        __builtin_amdgcn_s_wait_tensorcnt(0);
    }

    // --- Stage x slice (halo + zero boundary) as f16 (needs f32->f16 convert).
    {
        #pragma unroll
        for (int i = 0; i < 2; ++i) {
            const int j = tid + i * 256;                  // 0..511
            const int g = tbase - PADL + j;
            const float v = (g >= 0 && g < N_SAMPLES) ? x[(size_t)b * N_SAMPLES + g] : 0.0f;
            xs[j] = (_Float16)v;
        }
    }
    __syncthreads();

    const int wv   = tid >> 5;
    const int lane = tid & 31;
    const int mt   = wv & 3;            // M-tile (16 channels)
    const int ntb  = (wv >> 2) * 8;     // first N-tile of this wave
    const int m    = lane & 15;         // A: row M / B: col N / D: col N
    const int hh   = lane >> 4;         // lane half selector

    v8f acc[8] = {};

    union AFrag { v16h v; uint4 u[2]; };

    #pragma unroll
    for (int kc = 0; kc < 8; ++kc) {
        // A fragment: row M = mt*16+m, K chunk kc*32.
        // ISA 16-bit A layout: VGPR0-3 <- K = hh*8 + 0..7 ; VGPR4-7 <- K = 16 + hh*8 + 0..7
        AFrag af;
        const _Float16* wrow = lds_w + (mt * 16 + m) * W_STRIDE + kc * 32;
        af.u[0] = *(const uint4*)(wrow + hh * 8);         // 16B aligned
        af.u[1] = *(const uint4*)(wrow + 16 + hh * 8);    // 16B aligned

        #pragma unroll
        for (int i = 0; i < 8; ++i) {
            // B fragment: col N = m, K = kc*32 + hh*8 + e (e<8) / 16 + hh*8 + (e-8) (e>=8)  ->  xs[N + K]
            const _Float16* xp = xs + (ntb + i) * 16 + m + kc * 32 + hh * 8;
            v16h bf;
            #pragma unroll
            for (int e = 0; e < 16; ++e) bf[e] = xp[e + (e >= 8 ? 8 : 0)];

            acc[i] = __builtin_amdgcn_wmma_f32_16x16x32_f16(
                         /*neg_a=*/false, af.v, /*neg_b=*/false, bf,
                         /*c_mod=*/(short)0, acc[i],
                         /*reuse_a=*/false, /*reuse_b=*/false);
        }
    }

    // D layout: VGPR r -> M = hh*8 + r, N = m.
    #pragma unroll
    for (int i = 0; i < 8; ++i) {
        const int t = tbase + (ntb + i) * 16 + m;
        #pragma unroll
        for (int r = 0; r < 8; ++r) {
            const int ch = mt * 16 + hh * 8 + r;
            out[((size_t)b * N_OUT + ch) * N_SAMPLES + t] = acc[i][r];
        }
    }
}

// ---------------------------------------------------------------------------
extern "C" void kernel_launch(void* const* d_in, const int* in_sizes, int n_in,
                              void* d_out, int out_size, void* d_ws, size_t ws_size,
                              hipStream_t stream) {
    const float* x      = (const float*)d_in[0];   // (32, 1, 32768) f32
    const float* a      = (const float*)d_in[1];   // (64, 1) f32
    const float* mother = (const float*)d_in[2];   // (251,) f32
    _Float16* wbank = (_Float16*)d_ws;             // 64*256 f16 = 32 KB scratch
    float* out = (float*)d_out;                    // (32, 64, 32768) f32

    build_bank<<<N_OUT, K_PAD, 0, stream>>>(a, mother, wbank);

    dim3 grid(N_SAMPLES / TILE_T, BATCH);
    wavelet_conv_wmma<<<grid, 256, 0, stream>>>(x, wbank, out);
}
